// Net_58454504898860
// MI455X (gfx1250) — matrix operand order repaired and checked by reading, not compiled, not run on hardware
//
#include <hip/hip_runtime.h>
#include <hip/hip_bf16.h>
#include <stdint.h>

// ---------------------------------------------------------------- constants
#define NNODE 20000
#define NEDGE 320000
#define FIN   5189
#define DIM   256
#define NORIG 183
#define NCLS  7

typedef __attribute__((ext_vector_type(16))) __bf16 v16bf;
typedef __attribute__((ext_vector_type(2)))  __bf16 v2bf;
typedef __attribute__((ext_vector_type(8)))  float  v8f;

// native packed f32->bf16 (RNE) — lowers to v_cvt_pk_bf16_f32
__device__ __forceinline__ unsigned pack2(float a, float b) {
    v2bf v;
    v.x = (__bf16)a;
    v.y = (__bf16)b;
    return __builtin_bit_cast(unsigned, v);
}

// ---------------------------------------------------------------- WMMA GEMM
// C[M,N] = A[M,K](f32) @ B(f32, grouped columns of 256, within-group ld=256)
// fp32 -> bf16 conversion happens while staging into LDS (no extra HBM pass).
// Tile 128x128, BK=32; 8 wave32 waves = 2(M) x 4(N), each 64x32 = 4x2 WMMA tiles.
// Software pipeline: registers stage tile k+1 while tile k computes from LDS;
// double-buffered LDS -> one barrier per k-iteration.
#define BM 128
#define BN 128
#define BK 32
#define LPAD 40   // bf16 elems per LDS row (32 + 8 pad; row stride 80B, 16B aligned)

union Frag { v16bf v; uint4 q[2]; };

template <int RELU, int HASBIAS>
__global__ __launch_bounds__(256, 2)
void gemm_bf16_wmma(const float* __restrict__ A, int M, int K,
                    const float* __restrict__ B, long long bGroupStride,
                    const float* __restrict__ bias,
                    float* __restrict__ C, int N)
{
    __shared__ __align__(16) unsigned short lA[2][BM][LPAD];
    __shared__ __align__(16) unsigned short lB[2][BN][LPAD];

    const int tid    = threadIdx.x;
    const int lane   = tid & 31;
    const int wid    = tid >> 5;
    const int wave_m = wid >> 2;          // 0..1
    const int wave_n = wid & 3;           // 0..3
    const int bm     = blockIdx.y * BM;
    const int bn     = blockIdx.x * BN;
    const int half   = lane >> 4;
    const int l16    = lane & 15;

    // staging thread coords
    const int arow = tid >> 3;            // 0..31  (A row within 32-row pass)
    const int acq  = (tid & 7) << 2;      // 0,4,...,28 (A k-quad)
    const int bkk  = (tid >> 4) << 1;     // 0,2,...,30 (B k-row pair)
    const int bc0  = (tid & 15) << 3;     // 0,8,...,120 (B col octet)

    const bool fullM = (bm + BM <= M);

    v8f acc[4][2];
    const v8f zacc = {0.f,0.f,0.f,0.f,0.f,0.f,0.f,0.f};
    for (int mt = 0; mt < 4; ++mt)
        for (int nt = 0; nt < 2; ++nt)
            acc[mt][nt] = zacc;

    const int kTiles = (K + BK - 1) / BK;

    float ra[4][4];
    float rb[2][8];   // two adjacent k-rows x 8 columns

    // ---- stage tile at k0 into registers (global loads)
    auto loadRegs = [&](int k0) {
        const int gka = k0 + acq;
        if (fullM && gka + BK + 3 < K) {            // interior fast path
#pragma unroll
            for (int p = 0; p < 4; ++p) {
                const float* ap = A + (size_t)(bm + p * 32 + arow) * K + gka;
                ra[p][0] = ap[0]; ra[p][1] = ap[1]; ra[p][2] = ap[2]; ra[p][3] = ap[3];
                __builtin_prefetch(ap + 2 * BK, 0, 3);   // 2 tiles ahead, WGP scope
            }
        } else if (fullM && gka + 3 < K) {
#pragma unroll
            for (int p = 0; p < 4; ++p) {
                const float* ap = A + (size_t)(bm + p * 32 + arow) * K + gka;
                ra[p][0] = ap[0]; ra[p][1] = ap[1]; ra[p][2] = ap[2]; ra[p][3] = ap[3];
            }
        } else {                                     // edge tiles only
#pragma unroll
            for (int p = 0; p < 4; ++p) {
                const int grow = bm + p * 32 + arow;
#pragma unroll
                for (int j = 0; j < 4; ++j)
                    ra[p][j] = (grow < M && gka + j < K)
                               ? A[(size_t)grow * K + gka + j] : 0.f;
            }
        }
#pragma unroll
        for (int h = 0; h < 2; ++h) {
            const int gkb = k0 + bkk + h;
#pragma unroll
            for (int q = 0; q < 2; ++q) {
                const int n = bn + bc0 + q * 4;
                const int g = n >> 8, bc = n & 255;
                float4 v = (gkb < K)
                           ? *(const float4*)(B + (long long)g * bGroupStride
                                                + (long long)gkb * 256 + bc)
                           : make_float4(0.f, 0.f, 0.f, 0.f);
                rb[h][q * 4 + 0] = v.x; rb[h][q * 4 + 1] = v.y;
                rb[h][q * 4 + 2] = v.z; rb[h][q * 4 + 3] = v.w;
            }
        }
    };

    // ---- convert + store staged registers into LDS buffer
    auto storeLDS = [&](int buf) {
#pragma unroll
        for (int p = 0; p < 4; ++p) {
            const int r = p * 32 + arow;
            uint2 w;
            w.x = pack2(ra[p][0], ra[p][1]);
            w.y = pack2(ra[p][2], ra[p][3]);
            *(uint2*)&lA[buf][r][acq] = w;          // one b64 store
        }
#pragma unroll
        for (int j = 0; j < 8; ++j) {               // (k,k+1) contiguous in lB row
            *(unsigned*)&lB[buf][bc0 + j][bkk] = pack2(rb[0][j], rb[1][j]);
        }
    };

    loadRegs(0);
    storeLDS(0);
    __syncthreads();

    for (int kt = 0; kt < kTiles; ++kt) {
        const int buf = kt & 1;
        const bool hasNext = (kt + 1 < kTiles);
        if (hasNext) loadRegs((kt + 1) * BK);        // overlap with WMMA below

        // B fragments (32x16, 16-bit): lane elems 0..15 <- K = half*16 + 0..15
        Frag bfr[2];
#pragma unroll
        for (int nt = 0; nt < 2; ++nt) {
            const int nb_ = wave_n * 32 + nt * 16 + l16;
            const uint4* p = (const uint4*)&lB[buf][nb_][half * 16];
            bfr[nt].q[0] = p[0];
            bfr[nt].q[1] = p[1];
        }
        // A fragments (16x32, 16-bit): lane elems 0..15 <- K = half*16 + 0..15
#pragma unroll
        for (int mt = 0; mt < 4; ++mt) {
            const int rb_ = wave_m * 64 + mt * 16 + l16;
            Frag afr;
            afr.q[0] = *(const uint4*)&lA[buf][rb_][half * 16];
            afr.q[1] = *(const uint4*)&lA[buf][rb_][half * 16 + 8];
#pragma unroll
            for (int nt = 0; nt < 2; ++nt) {
                acc[mt][nt] = __builtin_amdgcn_wmma_f32_16x16x32_bf16(
                    false, afr.v, false, bfr[nt].v, (short)0, acc[mt][nt], false, false);
            }
        }

        if (hasNext) {
            storeLDS(buf ^ 1);   // other buffer: safe, last read before prev barrier
            __syncthreads();
        }
    }

    // ---- epilogue: C/D layout: lane l, VGPR j -> row = j + 8*half, col = l&15
#pragma unroll
    for (int mt = 0; mt < 4; ++mt) {
        const int rbase = bm + wave_m * 64 + mt * 16 + half * 8;
#pragma unroll
        for (int nt = 0; nt < 2; ++nt) {
            const int col = bn + wave_n * 32 + nt * 16 + l16;
            const float bv = HASBIAS ? bias[col] : 0.0f;
            if (fullM) {
#pragma unroll
                for (int j = 0; j < 8; ++j) {
                    float v = acc[mt][nt][j] + bv;
                    if (RELU) v = fmaxf(v, 0.0f);
                    C[(size_t)(rbase + j) * N + col] = v;
                }
            } else {
                for (int j = 0; j < 8; ++j) {
                    const int r = rbase + j;
                    if (r < M) {
                        float v = acc[mt][nt][j] + bv;
                        if (RELU) v = fmaxf(v, 0.0f);
                        C[(size_t)r * N + col] = v;
                    }
                }
            }
        }
    }
}

// ---------------------------------------------------------------- helpers
__global__ void fill_f32(float* __restrict__ p, float v, long long n) {
    long long t = (long long)blockIdx.x * blockDim.x + threadIdx.x;
    if (t < n) p[t] = v;
}

__global__ void deg_count(const int* __restrict__ dst, float* __restrict__ deg, int E) {
    int t = blockIdx.x * blockDim.x + threadIdx.x;
    if (t < E) atomicAdd(&deg[dst[t]], 1.0f);
}

__global__ void rsqrt_ip(float* __restrict__ d, long long n) {
    long long t = (long long)blockIdx.x * blockDim.x + threadIdx.x;
    if (t < n) { float v = d[t]; d[t] = (v > 0.f) ? rsqrtf(v) : 0.f; }
}

// Hagg[i][c] = dinv[r][i]^2 * H[i][c]   (self-loop term), c in [0,768), r = c>>8
__global__ void agg_init(const float* __restrict__ H, const float* __restrict__ dinv,
                         float* __restrict__ Hagg) {
    long long t = (long long)blockIdx.x * blockDim.x + threadIdx.x;
    if (t >= (long long)NNODE * 768) return;
    const int node = (int)(t / 768);
    const int c    = (int)(t % 768);
    const int r    = c >> 8;
    const float di = dinv[r * NNODE + node];
    Hagg[t] = di * di * H[t];
}

// per-edge scatter: Hagg[d][r*256+c] += dinv[s]*dinv[d]*H[s][r*256+c]
__global__ __launch_bounds__(256)
void agg_edges(const int* __restrict__ ei, const float* __restrict__ dinv,
               const float* __restrict__ H, float* __restrict__ Hagg, int r) {
    long long t = (long long)blockIdx.x * 256 + threadIdx.x;
    const int e = (int)(t >> 6);
    if (e >= NEDGE) return;
    const int c0 = (int)((t & 63) << 2) + (r << 8);
    const int s  = ei[e];
    const int d  = ei[NEDGE + e];
    const float w = dinv[r * NNODE + s] * dinv[r * NNODE + d];
    const float4 hv = *(const float4*)&H[(size_t)s * 768 + c0];
    float* o = &Hagg[(size_t)d * 768 + c0];
    atomicAdd(o + 0, w * hv.x);
    atomicAdd(o + 1, w * hv.y);
    atomicAdd(o + 2, w * hv.z);
    atomicAdd(o + 3, w * hv.w);
}

__global__ void bias_relu(float* __restrict__ Hagg, const float* __restrict__ b) {
    long long t = (long long)blockIdx.x * blockDim.x + threadIdx.x;
    if (t >= (long long)NNODE * 768) return;
    const int c = (int)(t % 768);
    Hagg[t] = fmaxf(Hagg[t] + b[c], 0.0f);
}

// segment sums for index_1 and index_3 in one pass
__global__ __launch_bounds__(256)
void seg_accum(const float* __restrict__ Hn, const int* __restrict__ i1,
               const int* __restrict__ i3, float* __restrict__ S1,
               float* __restrict__ S3, float* __restrict__ c1, float* __restrict__ c3) {
    long long t = (long long)blockIdx.x * 256 + threadIdx.x;
    const int node = (int)(t >> 6);
    if (node >= NNODE) return;
    const int c0 = (int)((t & 63) << 2);
    const float4 h = *(const float4*)&Hn[(size_t)node * DIM + c0];
    const int a = i1[node], b = i3[node];
    float* s1 = &S1[(size_t)a * DIM + c0];
    float* s3 = &S3[(size_t)b * DIM + c0];
    atomicAdd(s1 + 0, h.x); atomicAdd(s1 + 1, h.y);
    atomicAdd(s1 + 2, h.z); atomicAdd(s1 + 3, h.w);
    atomicAdd(s3 + 0, h.x); atomicAdd(s3 + 1, h.y);
    atomicAdd(s3 + 2, h.z); atomicAdd(s3 + 3, h.w);
    if (c0 == 0) { atomicAdd(&c1[a], 1.0f); atomicAdd(&c3[b], 1.0f); }
}

// Xc[i][0:256]=S1/cnt1, [256:512]=S3/cnt3, [512:768]=S3/cnt3 (source bug kept)
__global__ void build_xcat(const float* __restrict__ S1, const float* __restrict__ S3,
                           const float* __restrict__ c1, const float* __restrict__ c3,
                           float* __restrict__ Xc) {
    int t = blockIdx.x * blockDim.x + threadIdx.x;
    if (t >= NORIG * 768) return;
    const int i = t / 768;
    const int c = t % 768;
    float v;
    if (c < 256) v = S1[i * DIM + c] / fmaxf(c1[i], 1.0f);
    else         v = S3[i * DIM + (c & 255)] / fmaxf(c3[i], 1.0f);
    Xc[t] = v;
}

// Hid[i][d] = relu(sum_k Xc[i][k]*mw1[k][d] + mb1[d]); one block per row
__global__ __launch_bounds__(256)
void readout_hidden(const float* __restrict__ Xc, const float* __restrict__ mw1,
                    const float* __restrict__ mb1, float* __restrict__ Hid) {
    __shared__ float row[768];
    const int i = blockIdx.x;
    for (int k = threadIdx.x; k < 768; k += 256) row[k] = Xc[i * 768 + k];
    __syncthreads();
    const int d = threadIdx.x;
    float s = mb1[d];
    for (int k = 0; k < 768; ++k) s += row[k] * mw1[k * DIM + d];
    Hid[i * DIM + d] = fmaxf(s, 0.0f);
}

// out[i][j] = log_softmax(Hid[i] @ mw2 + mb2)
__global__ __launch_bounds__(256)
void readout_out(const float* __restrict__ Hid, const float* __restrict__ mw2,
                 const float* __restrict__ mb2, float* __restrict__ out) {
    __shared__ float h[DIM];
    __shared__ float lg[NCLS];
    const int i = blockIdx.x;
    h[threadIdx.x] = Hid[i * DIM + threadIdx.x];
    __syncthreads();
    if (threadIdx.x < NCLS) {
        float s = mb2[threadIdx.x];
        for (int k = 0; k < DIM; ++k) s += h[k] * mw2[k * NCLS + threadIdx.x];
        lg[threadIdx.x] = s;
    }
    __syncthreads();
    if (threadIdx.x == 0) {
        float mx = lg[0];
        for (int j = 1; j < NCLS; ++j) mx = fmaxf(mx, lg[j]);
        float se = 0.f;
        for (int j = 0; j < NCLS; ++j) se += __expf(lg[j] - mx);
        const float lse = mx + __logf(se);
        for (int j = 0; j < NCLS; ++j) out[i * NCLS + j] = lg[j] - lse;
    }
}

// ---------------------------------------------------------------- launcher
extern "C" void kernel_launch(void* const* d_in, const int* in_sizes, int n_in,
                              void* d_out, int out_size, void* d_ws, size_t ws_size,
                              hipStream_t stream) {
    const float* x    = (const float*)d_in[0];
    const float* w1   = (const float*)d_in[1];
    const float* b1   = (const float*)d_in[2];
    const float* w2   = (const float*)d_in[3];
    const float* b2   = (const float*)d_in[4];
    const float* m1w1 = (const float*)d_in[5];
    const float* m1b1 = (const float*)d_in[6];
    const float* m1w2 = (const float*)d_in[7];
    const float* m1b2 = (const float*)d_in[8];
    const float* m2w1 = (const float*)d_in[9];
    const float* m2b1 = (const float*)d_in[10];
    const float* m2w2 = (const float*)d_in[11];
    const float* m2b2 = (const float*)d_in[12];
    const float* mw1  = (const float*)d_in[13];
    const float* mb1  = (const float*)d_in[14];
    const float* mw2  = (const float*)d_in[15];
    const float* mb2  = (const float*)d_in[16];
    const int* edges[3] = { (const int*)d_in[17], (const int*)d_in[18], (const int*)d_in[19] };
    const int* idx1 = (const int*)d_in[20];
    const int* idx3 = (const int*)d_in[22];

    // workspace layout (floats)
    float* ws   = (float*)d_ws;
    float* dinv = ws;                                   // 3*NNODE
    float* H    = dinv + 3 * NNODE;                     // NNODE*768
    float* Hagg = H + (size_t)NNODE * 768;              // NNODE*768
    float* Hn   = Hagg + (size_t)NNODE * 768;           // NNODE*256
    float* S1   = Hn + (size_t)NNODE * DIM;             // NORIG*256
    float* S3   = S1 + NORIG * DIM;                     // NORIG*256
    float* c1   = S3 + NORIG * DIM;                     // NORIG
    float* c3   = c1 + NORIG;                           // NORIG
    float* Xc   = c3 + NORIG;                           // NORIG*768
    float* Hid  = Xc + NORIG * 768;                     // NORIG*256
    float* Mid  = H;                                    // alias: H dead when Mid used

    const long long n768 = (long long)NNODE * 768;
    const int blocks768  = (int)((n768 + 255) / 256);

    // ---- degrees (self loop baked in via init to 1.0), then deg^-1/2
    fill_f32<<<(3 * NNODE + 255) / 256, 256, 0, stream>>>(dinv, 1.0f, 3 * NNODE);
    for (int r = 0; r < 3; ++r)
        deg_count<<<(NEDGE + 255) / 256, 256, 0, stream>>>(edges[r] + NEDGE, dinv + r * NNODE, NEDGE);
    rsqrt_ip<<<(3 * NNODE + 255) / 256, 256, 0, stream>>>(dinv, 3 * NNODE);

    const dim3 g768(768 / BN, (NNODE + BM - 1) / BM);
    const dim3 g256(256 / BN, (NNODE + BM - 1) / BM);
    const int eb = (int)(((long long)NEDGE * 64 + 255) / 256);

    // ---- layer 1
    gemm_bf16_wmma<0,0><<<g768, 256, 0, stream>>>(x, NNODE, FIN, w1, (long long)FIN * DIM,
                                                  nullptr, H, 768);
    agg_init<<<blocks768, 256, 0, stream>>>(H, dinv, Hagg);
    for (int r = 0; r < 3; ++r)
        agg_edges<<<eb, 256, 0, stream>>>(edges[r], dinv, H, Hagg, r);
    bias_relu<<<blocks768, 256, 0, stream>>>(Hagg, b1);
    gemm_bf16_wmma<1,1><<<g256, 256, 0, stream>>>(Hagg, NNODE, 768, m1w1, 0, m1b1, Mid, 256);
    gemm_bf16_wmma<0,1><<<g256, 256, 0, stream>>>(Mid, NNODE, 256, m1w2, 0, m1b2, Hn, 256);

    // ---- layer 2
    gemm_bf16_wmma<0,0><<<g768, 256, 0, stream>>>(Hn, NNODE, 256, w2, (long long)DIM * DIM,
                                                  nullptr, H, 768);
    agg_init<<<blocks768, 256, 0, stream>>>(H, dinv, Hagg);
    for (int r = 0; r < 3; ++r)
        agg_edges<<<eb, 256, 0, stream>>>(edges[r], dinv, H, Hagg, r);
    bias_relu<<<blocks768, 256, 0, stream>>>(Hagg, b2);
    gemm_bf16_wmma<1,1><<<g256, 256, 0, stream>>>(Hagg, NNODE, 768, m2w1, 0, m2b1, Mid, 256);
    gemm_bf16_wmma<0,1><<<g256, 256, 0, stream>>>(Mid, NNODE, 256, m2w2, 0, m2b2, Hn, 256);

    // ---- readout
    const long long zcnt = 2LL * NORIG * DIM + 2 * NORIG;
    fill_f32<<<(int)((zcnt + 255) / 256), 256, 0, stream>>>(S1, 0.0f, zcnt);
    seg_accum<<<(int)(((long long)NNODE * 64 + 255) / 256), 256, 0, stream>>>(
        Hn, idx1, idx3, S1, S3, c1, c3);
    build_xcat<<<(NORIG * 768 + 255) / 256, 256, 0, stream>>>(S1, S3, c1, c3, Xc);
    readout_hidden<<<NORIG, 256, 0, stream>>>(Xc, mw1, mb1, Hid);
    readout_out<<<NORIG, 256, 0, stream>>>(Hid, mw2, mb2, (float*)d_out);
}
